// IaBNet_with_affinity_16458314678293
// MI455X (gfx1250) — matrix operand order repaired and checked by reading, not compiled, not run on hardware
//
#include <hip/hip_runtime.h>

// ---------------------------------------------------------------------------
// MI455X (gfx1250, wave32) implementation of the 5-layer trigonometry stack.
// Dense math: v_wmma_f32_16x16x32_bf16 (bf16 in, f32 acc).
// Data movement: GLOBAL_LOAD_ASYNC_TO_LDS_B128 (ASYNCcnt) double-buffered,
// with a VGPR-staging fallback when the async builtins are unavailable.
// ---------------------------------------------------------------------------

typedef __bf16 bf16;
typedef __attribute__((ext_vector_type(8)))  __bf16 v8bf;
typedef __attribute__((ext_vector_type(16))) __bf16 v16bf;
typedef __attribute__((ext_vector_type(8)))  float  v8f;
typedef __attribute__((ext_vector_type(4)))  float  v4f;

#define DEV __device__ __forceinline__

#if defined(__has_builtin)
#if __has_builtin(__builtin_amdgcn_global_load_async_to_lds_b128) && \
    __has_builtin(__builtin_amdgcn_s_wait_asynccnt)
#define ASYNC_COPY 1
#endif
#endif
#ifndef ASYNC_COPY
#define ASYNC_COPY 0
#endif

DEV v16bf cat8(v8bf a, v8bf b) {
  return __builtin_shufflevector(a, b, 0,1,2,3,4,5,6,7,8,9,10,11,12,13,14,15);
}
DEV float sigmoidf(float x) { return 1.0f / (1.0f + __expf(-x)); }
DEV void zero16(bf16* lp) { v4f z; z[0]=z[1]=z[2]=z[3]=0.0f; *(v4f*)lp = z; }

#if ASYNC_COPY
// Builtin prototype (from hipcc diagnostics): param0 = global int4*,
// param1 = LDS int4*, then imm offset, imm cpol.
typedef int v4i_vs __attribute__((vector_size(16)));
typedef __attribute__((address_space(1))) v4i_vs* gp128_t;
typedef __attribute__((address_space(3))) v4i_vs* lp128_t;
// 16B global -> LDS via the CDNA5 async copy engine (no VGPR round trip).
DEV void async_cp16(const bf16* gp, bf16* lp) {
  __builtin_amdgcn_global_load_async_to_lds_b128(
      (gp128_t)(gp), (lp128_t)(lp), 0, 0);
}
#endif

// ---------------------------------------------------------------------------
// General WMMA GEMM:  C = epilogue(A[M,K] @ Bt[N,K]^T)
//   A row-major (lda), Bt row-major [N][K] (ldb) -> contiguous K per lane.
//   Block tile 128x128, 8 waves, wave tile 32x64 (2x4 16x16 WMMA tiles).
//   K streamed through double-buffered LDS in chunks of KC = min(K,128).
// ---------------------------------------------------------------------------
struct GemmArgs {
  const bf16*  A;
  const bf16*  Bt;
  const float* bias;     // [N] or null
  const float* mul;      // elementwise multiplier (C indexing) or null
  const float* res;      // residual (C indexing) or null
  const float* rowmask;  // [M] or null
  float*       outF;     // f32 output or null
  bf16*        outH;     // bf16 output or null
  long sAB, sBB, sCB;    // batch strides (elements)
  int  M, N, K, lda, ldb;
  int  ldc, cstride;     // C index = b*sCB + ((long)row*ldc + col)*cstride
  int  act;              // 0 none, 1 relu, 2 sigmoid
};

#define TM 128
#define TN 128
#define TK 128

__global__ __launch_bounds__(256)
void k_gemm_bf16(GemmArgs g) {
  __shared__ __align__(16) bf16 sA[2][TM * TK];
  __shared__ __align__(16) bf16 sB[2][TN * TK];

  const int tid  = threadIdx.x;
  const int lane = tid & 31;
  const int wid  = tid >> 5;
  const int wm   = wid >> 1;      // 0..3 -> M offset wm*32
  const int wn   = wid & 1;       // 0..1 -> N offset wn*64
  const int lr   = lane & 15;
  const int hk   = lane >> 4;
  const int bm   = blockIdx.y * TM;
  const int bn   = blockIdx.x * TN;
  const long zb  = blockIdx.z;

  const bf16* A = g.A  + zb * g.sAB;
  const bf16* B = g.Bt + zb * g.sBB;

  v8f zero; for (int i = 0; i < 8; ++i) zero[i] = 0.0f;
  v8f acc[2][4];
  for (int m = 0; m < 2; ++m)
    for (int n = 0; n < 4; ++n) acc[m][n] = zero;

  const int KC    = (g.K < TK) ? g.K : TK;
  const int cw    = KC >> 3;           // 16B units per row of a chunk
  const int units = TM * cw;

#if ASYNC_COPY
  // Issue one chunk's A+B tiles as async b128 copies (tracked by ASYNCcnt).
  auto stage = [&](int buf, int kc) {
    for (int u = tid; u < units; u += 256) {
      int row = u / cw, c8 = u % cw;
      int grow = bm + row;
      bf16* lp = &sA[buf][row * TK + c8 * 8];
      if (grow < g.M) async_cp16(A + (long)grow * g.lda + kc + c8 * 8, lp);
      else            zero16(lp);
    }
    for (int u = tid; u < units; u += 256) {
      int row = u / cw, c8 = u % cw;
      int gn = bn + row;
      bf16* lp = &sB[buf][row * TK + c8 * 8];
      if (gn < g.N) async_cp16(B + (long)gn * g.ldb + kc + c8 * 8, lp);
      else          zero16(lp);
    }
  };
#else
  auto stage = [&](int buf, int kc) {
    for (int u = tid; u < units; u += 256) {
      int row = u / cw, c8 = u % cw;
      int grow = bm + row;
      v4f val; val[0]=val[1]=val[2]=val[3]=0.0f;
      if (grow < g.M) val = *(const v4f*)(A + (long)grow * g.lda + kc + c8 * 8);
      *(v4f*)(&sA[buf][row * TK + c8 * 8]) = val;
    }
    for (int u = tid; u < units; u += 256) {
      int row = u / cw, c8 = u % cw;
      int gn = bn + row;
      v4f val; val[0]=val[1]=val[2]=val[3]=0.0f;
      if (gn < g.N) val = *(const v4f*)(B + (long)gn * g.ldb + kc + c8 * 8);
      *(v4f*)(&sB[buf][row * TK + c8 * 8]) = val;
    }
  };
#endif

  auto compute = [&](int buf) {
    const bf16* cA = sA[buf];
    const bf16* cB = sB[buf];
    for (int k0 = 0; k0 < KC; k0 += 32) {
      v16bf aF[2], bF[4];
      #pragma unroll
      for (int m = 0; m < 2; ++m) {
        // A 16x32 bf16 fragment: lane<16 holds K {k0..+7, k0+16..+23}, row=lr
        const bf16* p = cA + (wm * 32 + m * 16 + lr) * TK + k0 + hk * 8;
        aF[m] = cat8(*(const v8bf*)p, *(const v8bf*)(p + 16));
      }
      #pragma unroll
      for (int n = 0; n < 4; ++n) {
        // B 32x16 bf16 fragment: lane<16 holds K {k0..+7, k0+16..+23}, col=lr
        const bf16* p = cB + (wn * 64 + n * 16 + lr) * TK + k0 + hk * 8;
        bF[n] = cat8(*(const v8bf*)p, *(const v8bf*)(p + 16));
      }
      #pragma unroll
      for (int m = 0; m < 2; ++m)
        #pragma unroll
        for (int n = 0; n < 4; ++n)
          acc[m][n] = __builtin_amdgcn_wmma_f32_16x16x32_bf16(
              false, aF[m], false, bF[n], (short)0, acc[m][n], false, false);
    }
  };

#if ASYNC_COPY
  // Double-buffered async pipeline: chunk i+1 streams in while chunk i is
  // consumed by the WMMA pipe.  s_wait_asynccnt 0 + barrier publishes a chunk.
  stage(0, 0);
  int buf = 0;
  for (int kc = 0; kc < g.K; kc += KC, buf ^= 1) {
    __builtin_amdgcn_s_wait_asynccnt(0);
    __syncthreads();
    if (kc + KC < g.K) {
      __builtin_prefetch(A + (long)(bm + (tid & 127)) * g.lda + kc + KC, 0, 1);
      stage(buf ^ 1, kc + KC);
    }
    compute(buf);
  }
#else
  for (int kc = 0; kc < g.K; kc += KC) {
    __syncthreads();
    if (kc + KC < g.K)
      __builtin_prefetch(A + (long)(bm + (tid & 127)) * g.lda + kc + KC, 0, 1);
    stage(0, kc);
    __syncthreads();
    compute(0);
  }
#endif

  // Epilogue: C layout -> VGPR r holds (M = r + hk*8, N = lr) of each 16x16 tile
  const long cb = zb * g.sCB;
  #pragma unroll
  for (int m = 0; m < 2; ++m) {
    #pragma unroll
    for (int n = 0; n < 4; ++n) {
      #pragma unroll
      for (int r = 0; r < 8; ++r) {
        int grow = bm + wm * 32 + m * 16 + hk * 8 + r;
        int gcol = bn + wn * 64 + n * 16 + lr;
        if (grow >= g.M || gcol >= g.N) continue;
        float v = acc[m][n][r];
        if (g.bias) v += g.bias[gcol];
        if (g.act == 1)      v = v > 0.0f ? v : 0.0f;
        else if (g.act == 2) v = sigmoidf(v);
        long idx = cb + ((long)grow * g.ldc + gcol) * (long)g.cstride;
        if (g.mul)     v *= g.mul[idx];
        if (g.rowmask) v *= g.rowmask[grow];
        if (g.res)     v += g.res[idx];
        if (g.outF) g.outF[idx] = v;
        if (g.outH) g.outH[idx] = (bf16)v;
      }
    }
  }
}

// ---------------------------------------------------------------------------
// Fused LayerNorm(x [+ x2]) -> f32 and/or bf16. One wave32 per row.
// ---------------------------------------------------------------------------
template <int W>
__global__ __launch_bounds__(256)
void k_ln(const float* __restrict__ x, const float* __restrict__ x2,
          const float* __restrict__ w, const float* __restrict__ b,
          float* __restrict__ outF, bf16* __restrict__ outH, long nrows) {
  constexpr int PL = W / 32;
  long row = (long)blockIdx.x * 8 + (threadIdx.x >> 5);
  if (row >= nrows) return;
  int lane = threadIdx.x & 31;
  float v[PL];
  float s = 0.0f;
  #pragma unroll
  for (int i = 0; i < PL; ++i) {
    long idx = row * W + lane + i * 32;
    float t = x[idx];
    if (x2) t += x2[idx];
    v[i] = t; s += t;
  }
  for (int off = 16; off; off >>= 1) s += __shfl_xor(s, off, 32);
  float mean = s / W;
  float vs = 0.0f;
  #pragma unroll
  for (int i = 0; i < PL; ++i) { float d = v[i] - mean; vs += d * d; }
  for (int off = 16; off; off >>= 1) vs += __shfl_xor(vs, off, 32);
  float rstd = rsqrtf(vs / W + 1e-5f);
  #pragma unroll
  for (int i = 0; i < PL; ++i) {
    int c = lane + i * 32;
    float o = (v[i] - mean) * rstd * w[c] + b[c];
    if (outF) outF[row * W + c] = o;
    if (outH) outH[row * W + c] = (bf16)o;
  }
}

// ---------------------------------------------------------------------------
// gate*linear (*rowmask) -> bf16, scattered channel-major for einsum operands.
// out[(long)c*R + t(r)] where t(r) = r (tW==0) or (r%tW)*tH + r/tW (transpose)
// ---------------------------------------------------------------------------
__global__ __launch_bounds__(256)
void k_gatemul(const float* __restrict__ g, const float* __restrict__ l,
               const float* __restrict__ rowmask, bf16* __restrict__ outT,
               long R, int Cc, int tW, long tH) {
  long i = (long)blockIdx.x * 256 + threadIdx.x;
  if (i >= R * Cc) return;
  long r = i / Cc; int c = (int)(i % Cc);
  float v = g[i] * l[i];
  if (rowmask) v *= rowmask[r];
  long tr = tW ? ((r % tW) * tH + r / tW) : r;
  outT[(long)c * R + tr] = (bf16)v;
}

// vec_dot reduction over NV=3: s[r2,c] = sum_K v1p*v2p  (vecp = [r2*3+K, 192])
__global__ __launch_bounds__(256)
void k_vecdot(const float* __restrict__ vp, bf16* __restrict__ out, long R2) {
  long i = (long)blockIdx.x * 256 + threadIdx.x;
  if (i >= R2 * 64) return;
  long r = i >> 6; int c = (int)(i & 63);
  const float* base = vp + r * 3 * 192;
  float s = 0.0f;
  for (int K = 0; K < 3; ++K) s += base[K * 192 + c] * base[K * 192 + 64 + c];
  out[i] = (bf16)s;
}

// ---------------------------------------------------------------------------
// Per-(row,head) attention: dd@dd_w (32x96), softmax over HC=32 lanes,
// emits x (bf16 -> o_w GEMM) and vv1/vv2 (f32).
// ---------------------------------------------------------------------------
__global__ __launch_bounds__(256)
void k_attn(const float* __restrict__ q, const float* __restrict__ kk,
            const float* __restrict__ znF, const float* __restrict__ vv,
            const float* __restrict__ ddw, const float* __restrict__ mask,
            bf16* __restrict__ xH, float* __restrict__ vv1,
            float* __restrict__ vv2, long R) {
  long rh = (long)blockIdx.x * 8 + (threadIdx.x >> 5);
  if (rh >= R * 4) return;
  long row = rh >> 2; int h = (int)(rh & 3);
  int lane = threadIdx.x & 31;
  int col  = h * 32 + lane;
  float dd = znF[row * 128 + col];
  float qv = q[row * 128 + col], kv = kk[row * 128 + col];
  float m0 = 0.f, m1 = 0.f, m2 = 0.f;
  for (int t = 0; t < 32; ++t) {
    float db = __shfl(dd, t, 32);
    const float* wr = ddw + t * 96;
    m0 += db * wr[lane]; m1 += db * wr[lane + 32]; m2 += db * wr[lane + 64];
  }
  const float* vr = vv + row * 384 + h * 96;
  float xv  = vr[lane]      * m0;
  float w1v = vr[lane + 32] * m1;
  float w2v = vr[lane + 64] * m2;
  float logit = qv * kv * dd + mask[row];
  float mx = logit;
  for (int off = 16; off; off >>= 1) mx = fmaxf(mx, __shfl_xor(mx, off, 32));
  float e = __expf(logit - mx), se = e;
  for (int off = 16; off; off >>= 1) se += __shfl_xor(se, off, 32);
  float wsm = e / se;
  xH [row * 128 + col] = (bf16)(wsm * xv);
  vv1[row * 128 + col] = w1v;
  vv2[row * 128 + col] = w2v;
}

// t = o1 * vec_dot * (1 + g2*l2*vv1 + g3*l3*vv2) -> bf16 (o1 strided in [R,384])
__global__ __launch_bounds__(256)
void k_tsacomb(const float* __restrict__ oF, const float* __restrict__ vd,
               const float* __restrict__ g2, const float* __restrict__ l2,
               const float* __restrict__ vv1, const float* __restrict__ g3,
               const float* __restrict__ l3, const float* __restrict__ vv2,
               bf16* __restrict__ out, long R) {
  long i = (long)blockIdx.x * 256 + threadIdx.x;
  if (i >= R * 128) return;
  long r = i >> 7; int c = (int)(i & 127);
  float o1  = oF[r * 384 + c];
  float oo2 = g2[i] * l2[i] * vv1[i];
  float oo3 = g3[i] * l3[i] * vv2[i];
  out[i] = (bf16)(o1 * vd[i] * (1.0f + oo2 + oo3));
}

// Weight transpose + cast: w[K,N] f32 -> out[N,K] bf16
__global__ __launch_bounds__(256)
void k_wt(const float* __restrict__ w, bf16* __restrict__ out, int K, int N) {
  long i = (long)blockIdx.x * 256 + threadIdx.x;
  if (i >= (long)K * N) return;
  int n = (int)(i / K), k = (int)(i % K);
  out[i] = (bf16)w[(long)k * N + n];
}

// ---------------------------------------------------------------------------
// Host orchestration
// ---------------------------------------------------------------------------
namespace {
enum { E = 128, C = 128, H = 4, HC = 32, NL = 5, NP = 512, NCc = 128, NV = 3, VD = 64 };
struct P2C { const float *eg_b,*eg_w,*g1_b,*g1_w,*g2_b,*g2_w,*l1_b,*l1_w,*l2_b,*l2_w,*las_b,*las_w,*ln_b,*ln_w,*lnc_b,*lnc_w; };
struct TR  { const float *b1,*b2,*ln_b,*ln_w,*w1,*w2; };
struct TSA { const float *dd_w,*fin_b,*fin_w,*g1_b,*g1_w,*g2_b,*g2_w,*k_w,*l1_b,*l1_w,*l2_b,*l2_w,*ln1_b,*ln1_w,*ln_b,*ln_w,*map128_w,*o_b,*o_w,*q_w,*v_w,*vec_proj_w; };
struct Layer { P2C p2c; TR tr; TSA tsa; };

static void gemm(hipStream_t st, const bf16* A, int lda, const bf16* Bt, int ldb,
                 int M, int N, int K, const float* bias, int act,
                 float* outF, bf16* outH,
                 const float* mul = nullptr, const float* res = nullptr,
                 const float* rowmask = nullptr,
                 int ldc = 0, int cstride = 1,
                 int batch = 1, long sAB = 0, long sBB = 0, long sCB = 0) {
  GemmArgs ga;
  ga.A = A; ga.Bt = Bt; ga.bias = bias; ga.mul = mul; ga.res = res; ga.rowmask = rowmask;
  ga.outF = outF; ga.outH = outH;
  ga.sAB = sAB; ga.sBB = sBB; ga.sCB = sCB;
  ga.M = M; ga.N = N; ga.K = K; ga.lda = lda; ga.ldb = ldb;
  ga.ldc = ldc ? ldc : N; ga.cstride = cstride; ga.act = act;
  dim3 grid((N + TN - 1) / TN, (M + TM - 1) / TM, batch);
  k_gemm_bf16<<<grid, dim3(256), 0, st>>>(ga);
}
static void wt(hipStream_t st, const float* w, bf16* dst, int K, int N) {
  long n = (long)K * N;
  k_wt<<<(unsigned)((n + 255) / 256), 256, 0, st>>>(w, dst, K, N);
}
} // namespace

extern "C" void kernel_launch(void* const* d_in, const int* in_sizes, int n_in,
                              void* d_out, int out_size, void* d_ws, size_t ws_size,
                              hipStream_t stream) {
  (void)in_sizes; (void)n_in; (void)out_size; (void)ws_size;
  const long R  = (long)NP * NCc;   // 65536 z rows
  const long Rp = (long)NP * NP;    // 262144 protein-pair rows
  const long Rc = (long)NCc * NCc;  // 16384 compound-pair rows
  const long Rv = R * NV;           // 196608 vec rows

  // d_in order = sorted top-level keys: compound_pair, params..., protein_pair, vec, z, z_mask
  const float* cp = (const float*)d_in[0];
  int pi = 1;
  auto nxt = [&]() { return (const float*)d_in[pi++]; };
  Layer L[NL];
  for (int l = 0; l < NL; ++l) {
    P2C& p = L[l].p2c;  // sorted keys within p2c dict
    p.eg_b=nxt(); p.eg_w=nxt(); p.g1_b=nxt(); p.g1_w=nxt(); p.g2_b=nxt(); p.g2_w=nxt();
    p.l1_b=nxt(); p.l1_w=nxt(); p.l2_b=nxt(); p.l2_w=nxt(); p.las_b=nxt(); p.las_w=nxt();
    p.ln_b=nxt(); p.ln_w=nxt(); p.lnc_b=nxt(); p.lnc_w=nxt();
    TR& t = L[l].tr;    // sorted: b1,b2,ln_b,ln_w,w1,w2
    t.b1=nxt(); t.b2=nxt(); t.ln_b=nxt(); t.ln_w=nxt(); t.w1=nxt(); t.w2=nxt();
    TSA& a = L[l].tsa;  // sorted keys within tsa dict
    a.dd_w=nxt(); a.fin_b=nxt(); a.fin_w=nxt(); a.g1_b=nxt(); a.g1_w=nxt();
    a.g2_b=nxt(); a.g2_w=nxt(); a.k_w=nxt(); a.l1_b=nxt(); a.l1_w=nxt();
    a.l2_b=nxt(); a.l2_w=nxt(); a.ln1_b=nxt(); a.ln1_w=nxt(); a.ln_b=nxt(); a.ln_w=nxt();
    a.map128_w=nxt(); a.o_b=nxt(); a.o_w=nxt(); a.q_w=nxt(); a.v_w=nxt(); a.vec_proj_w=nxt();
  }
  const float* pp   = (const float*)d_in[pi++];
  const float* vec  = (const float*)d_in[pi++];
  const float* zin  = (const float*)d_in[pi++];
  const float* mask = (const float*)d_in[pi++];

  // Workspace bump allocator (L2 is 192MB; z-sized tensors stay hot there).
  char* wp = (char*)d_ws;
  auto alloc = [&](size_t bytes) -> void* {
    void* p = wp; wp += (bytes + 255) & ~(size_t)255; return p;
  };
  auto aF = [&](long n) { return (float*)alloc((size_t)n * 4); };
  auto aH = [&](long n) { return (bf16*)alloc((size_t)n * 2); };

  float* zA   = aF(R * E);   float* zB   = aF(R * E);
  bf16*  znH  = aH(R * E);   float* znF  = aF(R * E);
  bf16*  ppnH = aH(Rp * E);  bf16*  cpnH = aH(Rc * E);
  float* gb   = aF(Rp * E);  float* lb   = aF(Rp * E);   // shared gate scratch
  bf16*  ab1T = aH((long)C * R);  bf16* ab2T = aH((long)C * R);
  bf16*  ppgT = aH((long)C * Rp); bf16* cpgT = aH((long)C * Rc);
  float* blk1 = aF(R * C);   float* blk2 = aF(R * C);
  bf16*  blnH = aH(R * C);   float* gateF = aF(R * E);
  float* qF = aF(R * E); float* kF = aF(R * E); float* vvF = aF(R * 3 * E);
  bf16*  vecnH = aH(Rv * VD); float* vecpF = aF(Rv * 3 * VD);
  bf16*  sH = aH(R * VD); float* vdF = aF(R * E);
  bf16*  xH = aH(R * E);  float* vv1F = aF(R * E); float* vv2F = aF(R * E);
  float* oF = aF(R * 3 * E); bf16* oH = aH(R * 3 * E);
  float* g2o = aF(R * E); float* l2o = aF(R * E);
  float* g3o = aF(R * E); float* l3o = aF(R * E);
  bf16*  tH  = aH(R * E); bf16* h1H = aH(R * 4 * E);
  bf16*  wt0 = aH(512 * 512); bf16* wt1 = aH(512 * 512);

  auto ln128 = [&](const float* x, const float* x2, const float* w, const float* b,
                   float* of, bf16* oh, long rows) {
    k_ln<128><<<(unsigned)((rows + 7) / 8), 256, 0, stream>>>(x, x2, w, b, of, oh, rows);
  };
  auto ln64 = [&](const float* x, const float* w, const float* b, bf16* oh, long rows) {
    k_ln<64><<<(unsigned)((rows + 7) / 8), 256, 0, stream>>>(x, nullptr, w, b, nullptr, oh, rows);
  };
  auto gml = [&](const float* g, const float* l, const float* m, bf16* o,
                 long rows, int tW, long tHt) {
    k_gatemul<<<(unsigned)((rows * C + 255) / 256), 256, 0, stream>>>(g, l, m, o, rows, C, tW, tHt);
  };

  const float* zcur = zin;
  for (int l = 0; l < NL; ++l) {
    const P2C& p = L[l].p2c; const TSA& a = L[l].tsa; const TR& t = L[l].tr;
    float* zout = (zcur == zA) ? zB : zA;

    // ---------------- TriangleProteinToCompound ----------------
    ln128(zcur, nullptr, p.ln_w, p.ln_b, nullptr, znH, R);
    ln128(pp,   nullptr, p.ln_w, p.ln_b, nullptr, ppnH, Rp);
    ln128(cp,   nullptr, p.ln_w, p.ln_b, nullptr, cpnH, Rc);
    // ab1 = sig(zn@g1)*(zn@l1)*mask  -> Bt of einsum1, scattered [c][j][k]
    wt(stream, p.g1_w, wt0, E, C); gemm(stream, znH, E, wt0, E, (int)R, C, E, p.g1_b, 2, gb, nullptr);
    wt(stream, p.l1_w, wt1, E, C); gemm(stream, znH, E, wt1, E, (int)R, C, E, p.l1_b, 0, lb, nullptr);
    gml(gb, lb, mask, ab1T, R, NCc, NP);
    // ab2 -> A of einsum2, direct [c][i][k]
    wt(stream, p.g2_w, wt0, E, C); gemm(stream, znH, E, wt0, E, (int)R, C, E, p.g2_b, 2, gb, nullptr);
    wt(stream, p.l2_w, wt1, E, C); gemm(stream, znH, E, wt1, E, (int)R, C, E, p.l2_b, 0, lb, nullptr);
    gml(gb, lb, mask, ab2T, R, 0, 0);
    // ppg -> A of einsum1 [c][i][k]
    wt(stream, p.g2_w, wt0, E, C); gemm(stream, ppnH, E, wt0, E, (int)Rp, C, E, p.g2_b, 2, gb, nullptr);
    wt(stream, p.l2_w, wt1, E, C); gemm(stream, ppnH, E, wt1, E, (int)Rp, C, E, p.l2_b, 0, lb, nullptr);
    gml(gb, lb, nullptr, ppgT, Rp, 0, 0);
    // cpg -> Bt of einsum2 [c][j][k]
    wt(stream, p.g1_w, wt0, E, C); gemm(stream, cpnH, E, wt0, E, (int)Rc, C, E, p.g1_b, 2, gb, nullptr);
    wt(stream, p.l1_w, wt1, E, C); gemm(stream, cpnH, E, wt1, E, (int)Rc, C, E, p.l1_b, 0, lb, nullptr);
    gml(gb, lb, nullptr, cpgT, Rc, 0, 0);
    // einsum1: per channel [512x512]@[512x128], out strided [i,j,c]
    gemm(stream, ppgT, NP, ab1T, NP, NP, NCc, NP, nullptr, 0, blk1, nullptr,
         nullptr, nullptr, nullptr, NCc, C, C, Rp, R, 1);
    // einsum2: per channel [512x128]@[128x128]^T
    gemm(stream, ab2T, NCc, cpgT, NCc, NP, NCc, NCc, nullptr, 0, blk2, nullptr,
         nullptr, nullptr, nullptr, NCc, C, C, R, Rc, 1);
    // gate, LN(block), final projection fused with gate*mask + residual
    wt(stream, p.eg_w, wt0, E, E); gemm(stream, znH, E, wt0, E, (int)R, E, E, p.eg_b, 2, gateF, nullptr);
    ln128(blk1, blk2, p.lnc_w, p.lnc_b, nullptr, blnH, R);
    wt(stream, p.las_w, wt1, C, E);
    gemm(stream, blnH, C, wt1, C, (int)R, E, C, p.las_b, 0, zout, nullptr, gateF, zcur, mask);
    zcur = zout; zout = (zcur == zA) ? zB : zA;

    // ---------------- TriangleSelfAttentionRowWise ----------------
    ln128(zcur, nullptr, a.ln_w, a.ln_b, znF, znH, R);
    wt(stream, a.q_w, wt0, E, H * HC); gemm(stream, znH, E, wt0, E, (int)R, H * HC, E, nullptr, 0, qF, nullptr);
    wt(stream, a.k_w, wt1, E, H * HC); gemm(stream, znH, E, wt1, E, (int)R, H * HC, E, nullptr, 0, kF, nullptr);
    wt(stream, a.v_w, wt0, E, 3 * H * HC); gemm(stream, znH, E, wt0, E, (int)R, 3 * H * HC, E, nullptr, 0, vvF, nullptr);
    ln64(vec, a.ln1_w, a.ln1_b, vecnH, Rv);
    wt(stream, a.vec_proj_w, wt1, VD, 3 * VD);
    gemm(stream, vecnH, VD, wt1, VD, (int)Rv, 3 * VD, VD, nullptr, 0, vecpF, nullptr);
    k_vecdot<<<(unsigned)((R * 64 + 255) / 256), 256, 0, stream>>>(vecpF, sH, R);
    wt(stream, a.map128_w, wt0, VD, E); gemm(stream, sH, VD, wt0, VD, (int)R, E, VD, nullptr, 0, vdF, nullptr);
    k_attn<<<(unsigned)((R * 4 + 7) / 8), 256, 0, stream>>>(qF, kF, znF, vvF, a.dd_w, mask, xH, vv1F, vv2F, R);
    wt(stream, a.o_w, wt1, E, 3 * E); gemm(stream, xH, E, wt1, E, (int)R, 3 * E, E, a.o_b, 0, oF, oH);
    wt(stream, a.g1_w, wt0, E, E); gemm(stream, oH + E,     3 * E, wt0, E, (int)R, E, E, a.g1_b, 2, g2o, nullptr);
    wt(stream, a.l1_w, wt1, E, E); gemm(stream, oH + E,     3 * E, wt1, E, (int)R, E, E, a.l1_b, 0, l2o, nullptr);
    wt(stream, a.g2_w, wt0, E, E); gemm(stream, oH + 2 * E, 3 * E, wt0, E, (int)R, E, E, a.g2_b, 2, g3o, nullptr);
    wt(stream, a.l2_w, wt1, E, E); gemm(stream, oH + 2 * E, 3 * E, wt1, E, (int)R, E, E, a.l2_b, 0, l3o, nullptr);
    k_tsacomb<<<(unsigned)((R * 128 + 255) / 256), 256, 0, stream>>>(oF, vdF, g2o, l2o, vv1F, g3o, l3o, vv2F, tH, R);
    wt(stream, a.fin_w, wt0, H * HC, E);
    gemm(stream, tH, E, wt0, E, (int)R, E, H * HC, a.fin_b, 0, zout, nullptr, nullptr, zcur, mask);
    zcur = zout; zout = (zcur == zA) ? zB : zA;

    // ---------------- Transition (E -> 4E -> E) ----------------
    ln128(zcur, nullptr, t.ln_w, t.ln_b, nullptr, znH, R);
    wt(stream, t.w1, wt1, E, 4 * E);
    gemm(stream, znH, E, wt1, E, (int)R, 4 * E, E, t.b1, 1, nullptr, h1H);
    wt(stream, t.w2, wt0, 4 * E, E);
    float* zfin = (l == NL - 1) ? (float*)d_out : zout;
    gemm(stream, h1H, 4 * E, wt0, 4 * E, (int)R, E, 4 * E, t.b2, 0, zfin, nullptr);
    zcur = zfin;
  }
}
